// EGNN_45208825758139
// MI455X (gfx1250) — compile-verified, matrix operand order checked by reading, not run on hardware
//
#include <hip/hip_runtime.h>
#include <hip/hip_bf16.h>
#include <math.h>

#define NN 20000
#define NE 160000
#define FIN 9
#define FF 64
#define NL 7
#define KPAD 160          // 2F+1 = 129 padded to 5*32
#define BN_EPS 1e-5f

typedef __attribute__((ext_vector_type(16))) _Float16 v16h;
typedef __attribute__((ext_vector_type(8)))  _Float16 v8h;
typedef __attribute__((ext_vector_type(4)))  _Float16 v4h;
typedef __attribute__((ext_vector_type(8)))  float    v8f;

// swizzled weight block offsets (halves), per layer
#define OFF_W1   0          // 5*4*512 = 10240
#define OFF_W2   10240      // 2*4*512 = 4096
#define OFF_PW1  14336      // 4096
#define OFF_PW2  18432      // 2*1*512 = 1024
#define OFF_UW1  19456      // 4*4*512 = 8192
#define OFF_UW2  27648      // 4096
#define OFF_UW3  31744      // 4096
#define WSW_PER_LAYER 35840
// global (non-layer) swizzled weights, appended after NL layers
#define GOFF_EMB1 0         // 1*4*512 = 2048
#define GOFF_EMB2 2048      // 4096
#define GOFF_EN1  6144      // 4096
#define GOFF_EN2  10240     // 4096
#define GOFF_EN3  14336     // 2*1*512 = 1024
#define WSW_GLOBAL 15360

// ---------------- WMMA helpers (gfx1250, wave32) ----------------

__device__ __forceinline__ v8f wmma16(v16h a, v16h b, v8f c) {
  return __builtin_amdgcn_wmma_f32_16x16x32_f16(
      false, a, false, b, (short)0, c, false, false);
}

// A fragment: 16x32 f16, rows = lane&15, from LDS row-major [16][row_stride]
__device__ __forceinline__ v16h load_a_frag(const _Float16* lds, int row_stride,
                                            int kb, int lane) {
  const _Float16* p = lds + (lane & 15) * row_stride + kb * 32 + ((lane & 16) ? 8 : 0);
  v8h lo = *(const v8h*)p;          // 16B aligned
  v8h hi = *(const v8h*)(p + 16);   // 16B aligned
  return __builtin_shufflevector(lo, hi, 0, 1, 2, 3, 4, 5, 6, 7,
                                 8, 9, 10, 11, 12, 13, 14, 15);
}

// B fragment from pre-swizzled LDS: [tile][lane][16 halves] contiguous
__device__ __forceinline__ v16h load_b_frag_sw(const _Float16* w, int numNt,
                                               int kb, int nt, int lane) {
  return *(const v16h*)(w + (((kb * numNt + nt) << 5) + lane) * 16);
}

// fast silu: x * rcp(1+exp(-x)) — v_exp_f32 + v_rcp_f32, no div-fixup sequence
__device__ __forceinline__ float silu(float x) {
  return x * __builtin_amdgcn_rcpf(1.f + __expf(-x));
}

// contiguous f16 copy global->LDS, 8 halves (16B) per op
__device__ __forceinline__ void copy16(const _Float16* __restrict__ g,
                                       _Float16* __restrict__ s,
                                       int halves, int tid, int nth) {
  const int4* gi = (const int4*)g;
  int4* si = (int4*)s;
  int n = halves >> 3;
  for (int i = tid; i < n; i += nth) si[i] = gi[i];
}

// ---------------- weight pre-swizzle ----------------

__device__ void swizzleB(const float* __restrict__ W, int kdim, int ncols, int ldw,
                         int numKb, int numNt, _Float16* __restrict__ dst,
                         int tid, int nth) {
  int total = numKb * numNt * 512;
  for (int idx = tid; idx < total; idx += nth) {
    int i    = idx & 15;
    int lane = (idx >> 4) & 31;
    int tile = idx >> 9;
    int kb = tile / numNt, nt = tile - kb * numNt;
    int k = kb * 32 + ((lane & 16) ? 8 : 0) + ((i & 8) ? 16 : 0) + (i & 7);
    int n = nt * 16 + (lane & 15);
    float v = (k < kdim && n < ncols) ? W[k * ldw + n] : 0.f;
    dst[idx] = (_Float16)v;
  }
}

__global__ void prep_weights_kernel(const float* __restrict__ msg_W1,
                                    const float* __restrict__ msg_W2,
                                    const float* __restrict__ pos_W1,
                                    const float* __restrict__ pos_W2,
                                    const float* __restrict__ upd_W1,
                                    const float* __restrict__ upd_W2,
                                    const float* __restrict__ upd_W3,
                                    const float* __restrict__ emb_W1,
                                    const float* __restrict__ emb_W2,
                                    const float* __restrict__ en_W1,
                                    const float* __restrict__ en_W2,
                                    const float* __restrict__ en_W3,
                                    _Float16* __restrict__ wsw) {
  const int l = blockIdx.x;
  const int tid = threadIdx.x, nth = blockDim.x;
  if (l < NL) {
    _Float16* p = wsw + (size_t)l * WSW_PER_LAYER;
    swizzleB(msg_W1 + (size_t)l * (2 * FF + 1) * FF, 2 * FF + 1, FF, FF, 5, 4, p + OFF_W1, tid, nth);
    swizzleB(msg_W2 + (size_t)l * FF * FF, FF, FF, FF, 2, 4, p + OFF_W2, tid, nth);
    swizzleB(pos_W1 + (size_t)l * FF * FF, FF, FF, FF, 2, 4, p + OFF_PW1, tid, nth);
    swizzleB(pos_W2 + (size_t)l * FF * 3, FF, 3, 3, 2, 1, p + OFF_PW2, tid, nth);
    swizzleB(upd_W1 + (size_t)l * 2 * FF * FF, 2 * FF, FF, FF, 4, 4, p + OFF_UW1, tid, nth);
    swizzleB(upd_W2 + (size_t)l * FF * FF, FF, FF, FF, 2, 4, p + OFF_UW2, tid, nth);
    swizzleB(upd_W3 + (size_t)l * FF * FF, FF, FF, FF, 2, 4, p + OFF_UW3, tid, nth);
  } else {
    _Float16* g = wsw + (size_t)NL * WSW_PER_LAYER;
    swizzleB(emb_W1, FIN, FF, FF, 1, 4, g + GOFF_EMB1, tid, nth);
    swizzleB(emb_W2, FF, FF, FF, 2, 4, g + GOFF_EMB2, tid, nth);
    swizzleB(en_W1, FF, FF, FF, 2, 4, g + GOFF_EN1, tid, nth);
    swizzleB(en_W2, FF, FF, FF, 2, 4, g + GOFF_EN2, tid, nth);
    swizzleB(en_W3, FF, 1, 1, 2, 1, g + GOFF_EN3, tid, nth);
  }
}

// ---------------- small kernels ----------------

__global__ void count_kernel(const int* __restrict__ dst, float* __restrict__ cnt) {
  int e = blockIdx.x * blockDim.x + threadIdx.x;
  if (e < NE) atomicAdd(&cnt[dst[e]], 1.0f);
}

__global__ void bn_finalize_kernel(const float* __restrict__ sum,
                                   const float* __restrict__ sq,
                                   const float* __restrict__ gamma,
                                   const float* __restrict__ beta,
                                   float* __restrict__ scale,
                                   float* __restrict__ shift) {
  int c = threadIdx.x;
  if (c >= FF) return;
  const float inv = 1.0f / (float)NE;
  float mu  = sum[c] * inv;
  float var = sq[c] * inv - mu * mu;
  float sc  = gamma[c] * rsqrtf(var + BN_EPS);
  scale[c] = sc;
  shift[c] = beta[c] - mu * sc;
}

// ---------------- embed (WMMA): h = silu(h_in@W1+b1)@W2+b2 ----------------
// grid: ceil(NN/64) blocks, 4 waves x 16-node tiles

__global__ __launch_bounds__(128) void embed_kernel(
    const float* __restrict__ h_in, const _Float16* __restrict__ wswG,
    const float* __restrict__ b1, const float* __restrict__ b2,
    float* __restrict__ h) {
  __shared__ __align__(64) _Float16 wts[6144];          // embW1 | embW2 swizzled
  __shared__ __align__(64) _Float16 feat[4][16 * FF];
  const int tid = threadIdx.x, wave = tid >> 5, lane = tid & 31;

  copy16(wswG + GOFF_EMB1, wts, 6144, tid, 128);
  const _Float16* w1s = wts;          // 1 kb, 4 nt
  const _Float16* w2s = wts + 2048;   // 2 kb, 4 nt
  __syncthreads();

  const int tileBase = (blockIdx.x * 4 + wave) * 16;
  _Float16* ft = feat[wave];
  const int hi8 = (lane & 16) ? 8 : 0;

  // fill cols 0..31 (FIN=9 valid), row stride 64
  for (int idx = lane; idx < 16 * 32; idx += 32) {
    int r = idx >> 5, c = idx & 31;
    int i = tileBase + r;
    float v = (i < NN && c < FIN) ? h_in[(size_t)i * FIN + c] : 0.f;
    ft[r * FF + c] = (_Float16)v;
  }

  v8f acc[4] = {v8f{}, v8f{}, v8f{}, v8f{}};
  {
    v16h a = load_a_frag(ft, FF, 0, lane);
#pragma unroll
    for (int nt = 0; nt < 4; ++nt)
      acc[nt] = wmma16(a, load_b_frag_sw(w1s, 4, 0, nt, lane), acc[nt]);
  }
#pragma unroll
  for (int nt = 0; nt < 4; ++nt) {
    const int col = (lane & 15) + nt * 16;
    const float bias = b1[col];
#pragma unroll
    for (int r = 0; r < 8; ++r)
      ft[(r + hi8) * FF + col] = (_Float16)silu(acc[nt][r] + bias);
  }

  v8f acc2[4] = {v8f{}, v8f{}, v8f{}, v8f{}};
#pragma unroll
  for (int kb = 0; kb < 2; ++kb) {
    v16h a = load_a_frag(ft, FF, kb, lane);
#pragma unroll
    for (int nt = 0; nt < 4; ++nt)
      acc2[nt] = wmma16(a, load_b_frag_sw(w2s, 4, kb, nt, lane), acc2[nt]);
  }
#pragma unroll
  for (int nt = 0; nt < 4; ++nt) {
    const int col = (lane & 15) + nt * 16;
    const float bias = b2[col];
#pragma unroll
    for (int r = 0; r < 8; ++r) {
      int i = tileBase + r + hi8;
      if (i < NN) h[(size_t)i * FF + col] = acc2[nt][r] + bias;
    }
  }
}

// ---------------- energy (WMMA): out = silu(silu(h@W1+b1)@W2+b2)@W3+b3 ----------------

__global__ __launch_bounds__(128) void energy_kernel(
    const float* __restrict__ h, const _Float16* __restrict__ wswG,
    const float* __restrict__ b1, const float* __restrict__ b2,
    const float* __restrict__ b3, float* __restrict__ out) {
  __shared__ __align__(64) _Float16 wts[9216];          // enW1 | enW2 | enW3pad
  __shared__ __align__(64) _Float16 feat[4][16 * FF];
  const int tid = threadIdx.x, wave = tid >> 5, lane = tid & 31;

  copy16(wswG + GOFF_EN1, wts, 9216, tid, 128);
  const _Float16* w1s = wts;
  const _Float16* w2s = wts + 4096;
  const _Float16* w3s = wts + 8192;
  __syncthreads();

  const int tileBase = (blockIdx.x * 4 + wave) * 16;
  _Float16* ft = feat[wave];
  const int hi8 = (lane & 16) ? 8 : 0;

  for (int g = lane; g < 16 * (FF / 4); g += 32) {
    int r = g >> 4, c4 = (g & 15) * 4;
    int i = tileBase + r;
    v4h o = {};
    if (i < NN) {
      float4 v = *(const float4*)(h + (size_t)i * FF + c4);
      o[0] = (_Float16)v.x; o[1] = (_Float16)v.y;
      o[2] = (_Float16)v.z; o[3] = (_Float16)v.w;
    }
    *(v4h*)(ft + r * FF + c4) = o;
  }

  v8f acc[4] = {v8f{}, v8f{}, v8f{}, v8f{}};
#pragma unroll
  for (int kb = 0; kb < 2; ++kb) {
    v16h a = load_a_frag(ft, FF, kb, lane);
#pragma unroll
    for (int nt = 0; nt < 4; ++nt)
      acc[nt] = wmma16(a, load_b_frag_sw(w1s, 4, kb, nt, lane), acc[nt]);
  }
#pragma unroll
  for (int nt = 0; nt < 4; ++nt) {
    const int col = (lane & 15) + nt * 16;
    const float bias = b1[col];
#pragma unroll
    for (int r = 0; r < 8; ++r)
      ft[(r + hi8) * FF + col] = (_Float16)silu(acc[nt][r] + bias);
  }

  v8f acc2[4] = {v8f{}, v8f{}, v8f{}, v8f{}};
#pragma unroll
  for (int kb = 0; kb < 2; ++kb) {
    v16h a = load_a_frag(ft, FF, kb, lane);
#pragma unroll
    for (int nt = 0; nt < 4; ++nt)
      acc2[nt] = wmma16(a, load_b_frag_sw(w2s, 4, kb, nt, lane), acc2[nt]);
  }
#pragma unroll
  for (int nt = 0; nt < 4; ++nt) {
    const int col = (lane & 15) + nt * 16;
    const float bias = b2[col];
#pragma unroll
    for (int r = 0; r < 8; ++r)
      ft[(r + hi8) * FF + col] = (_Float16)silu(acc2[nt][r] + bias);
  }

  v8f acc3 = v8f{};
#pragma unroll
  for (int kb = 0; kb < 2; ++kb) {
    v16h a = load_a_frag(ft, FF, kb, lane);
    acc3 = wmma16(a, load_b_frag_sw(w3s, 1, kb, 0, lane), acc3);
  }
  if ((lane & 15) == 0) {
    const float bias = b3[0];
#pragma unroll
    for (int r = 0; r < 8; ++r) {
      int i = tileBase + r + hi8;
      if (i < NN) out[i] = acc3[r] + bias;
    }
  }
}

// ---------------- edge message stage 1 (32-edge tiles per wave) ----------------
// grid: NE/128 blocks, 128 threads (4 waves x 32-edge tiles)

__global__ __launch_bounds__(128) void edge_msg1_kernel(
    const float* __restrict__ h, const float* __restrict__ pos,
    const float* __restrict__ cell,
    const int* __restrict__ src, const int* __restrict__ dst,
    const _Float16* __restrict__ wsw, const float* __restrict__ b1,
    float* __restrict__ m1, float* __restrict__ bnSum, float* __restrict__ bnSq) {
  __shared__ __align__(64) _Float16 w1s[5 * 4 * 512];     // 20480 B swizzled
  __shared__ __align__(64) _Float16 feat[4][32 * KPAD];   // 4 x 10240 B
  const int tid = threadIdx.x, wave = tid >> 5, lane = tid & 31;

  copy16(wsw + OFF_W1, w1s, 5 * 4 * 512, tid, 128);
  __syncthreads();

  const int tileBase = (blockIdx.x * 4 + wave) * 32;
  _Float16* ft = feat[wave];

  // build feat tile [32][160] = [h_dst | h_src | dist | 0...]
  for (int g = lane; g < 32 * (KPAD / 4); g += 32) {
    int r = g / (KPAD / 4), c4 = (g - r * (KPAD / 4)) * 4;
    int e = tileBase + r;
    v4h o = {};
    if (e < NE) {
      int s = src[e], d = dst[e];
      if (c4 < FF) {
        float4 v = *(const float4*)(h + (size_t)d * FF + c4);
        o[0] = (_Float16)v.x; o[1] = (_Float16)v.y;
        o[2] = (_Float16)v.z; o[3] = (_Float16)v.w;
      } else if (c4 < 2 * FF) {
        float4 v = *(const float4*)(h + (size_t)s * FF + (c4 - FF));
        o[0] = (_Float16)v.x; o[1] = (_Float16)v.y;
        o[2] = (_Float16)v.z; o[3] = (_Float16)v.w;
      } else if (c4 == 2 * FF) {
        float dx = pos[d * 3 + 0] - pos[s * 3 + 0] + cell[e * 3 + 0];
        float dy = pos[d * 3 + 1] - pos[s * 3 + 1] + cell[e * 3 + 1];
        float dz = pos[d * 3 + 2] - pos[s * 3 + 2] + cell[e * 3 + 2];
        o[0] = (_Float16)(dx + dy + dz);
      }
    }
    *(v4h*)(ft + r * KPAD + c4) = o;
  }

  v8f accL[4] = {v8f{}, v8f{}, v8f{}, v8f{}};
  v8f accH[4] = {v8f{}, v8f{}, v8f{}, v8f{}};
#pragma unroll
  for (int kb = 0; kb < 5; ++kb) {
    v16h aL = load_a_frag(ft, KPAD, kb, lane);
    v16h aH = load_a_frag(ft + 16 * KPAD, KPAD, kb, lane);
#pragma unroll
    for (int nt = 0; nt < 4; ++nt) {
      v16h b = load_b_frag_sw(w1s, 4, kb, nt, lane);
      accL[nt] = wmma16(aL, b, accL[nt]);
      accH[nt] = wmma16(aH, b, accH[nt]);
    }
  }

  const int hi8 = (lane & 16) ? 8 : 0;
#pragma unroll
  for (int nt = 0; nt < 4; ++nt) {
    const int col = (lane & 15) + nt * 16;
    const float bias = b1[col];
    float s1 = 0.f, s2 = 0.f;
#pragma unroll
    for (int r = 0; r < 8; ++r) {
      int e = tileBase + r + hi8;
      float x = silu(accL[nt][r] + bias);
      if (e < NE) { m1[(size_t)e * FF + col] = x; s1 += x; s2 += x * x; }
      int e2 = e + 16;
      float y = silu(accH[nt][r] + bias);
      if (e2 < NE) { m1[(size_t)e2 * FF + col] = y; s1 += y; s2 += y * y; }
    }
    atomicAdd(&bnSum[col], s1);
    atomicAdd(&bnSq[col], s2);
  }
}

// ---------------- edge message stage 2 (32-edge tiles per wave) ----------------

__global__ __launch_bounds__(128) void edge_msg2_kernel(
    const float* __restrict__ m1,
    const float* __restrict__ bnScale, const float* __restrict__ bnShift,
    const _Float16* __restrict__ wsw,
    const float* __restrict__ b2, const float* __restrict__ pb1,
    const float* __restrict__ pb2,
    const float* __restrict__ pos, const float* __restrict__ cell,
    const int* __restrict__ src, const int* __restrict__ dst,
    float* __restrict__ aggm, float* __restrict__ aggpm) {
  __shared__ __align__(64) _Float16 wts[9216];        // W2 | pW1 | pW2pad swizzled
  __shared__ __align__(64) _Float16 feat[4][32 * FF]; // 4 x 4096 B
  const int tid = threadIdx.x, wave = tid >> 5, lane = tid & 31;

  copy16(wsw + OFF_W2, wts, 9216, tid, 128);   // contiguous in prep layout
  const _Float16* w2s  = wts;
  const _Float16* pw1s = wts + 4096;
  const _Float16* pw2s = wts + 8192;
  __syncthreads();

  const int tileBase = (blockIdx.x * 4 + wave) * 32;
  _Float16* ft = feat[wave];
  const int hi8 = (lane & 16) ? 8 : 0;

  // m1 tile + BN affine -> f16
  for (int g = lane; g < 32 * (FF / 4); g += 32) {
    int r = g >> 4, c4 = (g & 15) * 4;
    int e = tileBase + r;
    v4h o = {};
    if (e < NE) {
      float4 v = *(const float4*)(m1 + (size_t)e * FF + c4);
      o[0] = (_Float16)(v.x * bnScale[c4 + 0] + bnShift[c4 + 0]);
      o[1] = (_Float16)(v.y * bnScale[c4 + 1] + bnShift[c4 + 1]);
      o[2] = (_Float16)(v.z * bnScale[c4 + 2] + bnShift[c4 + 2]);
      o[3] = (_Float16)(v.w * bnScale[c4 + 3] + bnShift[c4 + 3]);
    }
    *(v4h*)(ft + r * FF + c4) = o;
  }

  // m = mA @ W2 + b2
  v8f mL[4] = {v8f{}, v8f{}, v8f{}, v8f{}};
  v8f mH[4] = {v8f{}, v8f{}, v8f{}, v8f{}};
#pragma unroll
  for (int kb = 0; kb < 2; ++kb) {
    v16h aL = load_a_frag(ft, FF, kb, lane);
    v16h aH = load_a_frag(ft + 16 * FF, FF, kb, lane);
#pragma unroll
    for (int nt = 0; nt < 4; ++nt) {
      v16h b = load_b_frag_sw(w2s, 4, kb, nt, lane);
      mL[nt] = wmma16(aL, b, mL[nt]);
      mH[nt] = wmma16(aH, b, mH[nt]);
    }
  }
#pragma unroll
  for (int nt = 0; nt < 4; ++nt) {
    const int col = (lane & 15) + nt * 16;
    const float bias = b2[col];
#pragma unroll
    for (int r = 0; r < 8; ++r) {
      float x = mL[nt][r] + bias;
      mL[nt][r] = x;
      ft[(r + hi8) * FF + col] = (_Float16)x;
      float y = mH[nt][r] + bias;
      mH[nt][r] = y;
      ft[(16 + r + hi8) * FF + col] = (_Float16)y;
    }
  }

  // scatter agg_m now so the m accumulators die before the pos-MLP ones
#pragma unroll
  for (int nt = 0; nt < 4; ++nt) {
    const int col = (lane & 15) + nt * 16;
#pragma unroll
    for (int r = 0; r < 8; ++r) {
      int e = tileBase + r + hi8;
      if (e < NE) atomicAdd(&aggm[(size_t)dst[e] * FF + col], mL[nt][r]);
      int e2 = e + 16;
      if (e2 < NE) atomicAdd(&aggm[(size_t)dst[e2] * FF + col], mH[nt][r]);
    }
  }

  // t = silu(m @ pW1 + pb1)
  v8f tL[4] = {v8f{}, v8f{}, v8f{}, v8f{}};
  v8f tH[4] = {v8f{}, v8f{}, v8f{}, v8f{}};
#pragma unroll
  for (int kb = 0; kb < 2; ++kb) {
    v16h aL = load_a_frag(ft, FF, kb, lane);
    v16h aH = load_a_frag(ft + 16 * FF, FF, kb, lane);
#pragma unroll
    for (int nt = 0; nt < 4; ++nt) {
      v16h b = load_b_frag_sw(pw1s, 4, kb, nt, lane);
      tL[nt] = wmma16(aL, b, tL[nt]);
      tH[nt] = wmma16(aH, b, tH[nt]);
    }
  }
#pragma unroll
  for (int nt = 0; nt < 4; ++nt) {
    const int col = (lane & 15) + nt * 16;
    const float bias = pb1[col];
#pragma unroll
    for (int r = 0; r < 8; ++r) {
      ft[(r + hi8) * FF + col]      = (_Float16)silu(tL[nt][r] + bias);
      ft[(16 + r + hi8) * FF + col] = (_Float16)silu(tH[nt][r] + bias);
    }
  }

  // s = t @ pW2pad  (cols 0..2 valid)
  v8f sL = v8f{}, sH = v8f{};
#pragma unroll
  for (int kb = 0; kb < 2; ++kb) {
    v16h bq = load_b_frag_sw(pw2s, 1, kb, 0, lane);
    sL = wmma16(load_a_frag(ft, FF, kb, lane), bq, sL);
    sH = wmma16(load_a_frag(ft + 16 * FF, FF, kb, lane), bq, sH);
  }

  const int c3 = lane & 15;
  if (c3 < 3) {
    const float bias = pb2[c3];
#pragma unroll
    for (int r = 0; r < 8; ++r) {
      int e = tileBase + r + hi8;
      if (e < NE) {
        int s = src[e], d = dst[e];
        float dv = pos[d * 3 + c3] - pos[s * 3 + c3] + cell[e * 3 + c3];
        atomicAdd(&aggpm[d * 3 + c3], dv * (sL[r] + bias));
      }
      int e2 = e + 16;
      if (e2 < NE) {
        int s = src[e2], d = dst[e2];
        float dv = pos[d * 3 + c3] - pos[s * 3 + c3] + cell[e2 * 3 + c3];
        atomicAdd(&aggpm[d * 3 + c3], dv * (sH[r] + bias));
      }
    }
  }
}

// ---------------- node update ----------------

__global__ __launch_bounds__(128) void node_update_kernel(
    float* __restrict__ h, float* __restrict__ posw,
    const float* __restrict__ aggm, const float* __restrict__ aggpm,
    const float* __restrict__ cnt,
    const _Float16* __restrict__ wsw,
    const float* __restrict__ ub1, const float* __restrict__ ub2,
    const float* __restrict__ ub3) {
  __shared__ __align__(64) _Float16 wts[16384];           // uW1|uW2|uW3 swizzled
  __shared__ __align__(64) _Float16 feat[4][16 * 2 * FF]; // 4 x 4096 B
  const int tid = threadIdx.x, wave = tid >> 5, lane = tid & 31;

  copy16(wsw + OFF_UW1, wts, 16384, tid, 128);
  const _Float16* uw1s = wts;
  const _Float16* uw2s = wts + 8192;
  const _Float16* uw3s = wts + 12288;
  __syncthreads();

  const int tileBase = (blockIdx.x * 4 + wave) * 16;
  _Float16* ft = feat[wave];
  const int hi8 = (lane & 16) ? 8 : 0;

  // u = [h | agg_m / cnt]   (16 x 128)
  for (int g = lane; g < 16 * (2 * FF / 4); g += 32) {
    int r = g >> 5, c4 = (g & 31) * 4;
    int i = tileBase + r;
    v4h o = {};
    if (i < NN) {
      if (c4 < FF) {
        float4 v = *(const float4*)(h + (size_t)i * FF + c4);
        o[0] = (_Float16)v.x; o[1] = (_Float16)v.y;
        o[2] = (_Float16)v.z; o[3] = (_Float16)v.w;
      } else {
        float rc = __builtin_amdgcn_rcpf(fmaxf(cnt[i], 1.f));
        float4 v = *(const float4*)(aggm + (size_t)i * FF + (c4 - FF));
        o[0] = (_Float16)(v.x * rc); o[1] = (_Float16)(v.y * rc);
        o[2] = (_Float16)(v.z * rc); o[3] = (_Float16)(v.w * rc);
      }
    }
    *(v4h*)(ft + r * 2 * FF + c4) = o;
  }

  // u1 = silu(u @ uW1 + ub1)
  v8f acc[4] = {v8f{}, v8f{}, v8f{}, v8f{}};
#pragma unroll
  for (int kb = 0; kb < 4; ++kb) {
    v16h a = load_a_frag(ft, 2 * FF, kb, lane);
#pragma unroll
    for (int nt = 0; nt < 4; ++nt)
      acc[nt] = wmma16(a, load_b_frag_sw(uw1s, 4, kb, nt, lane), acc[nt]);
  }
#pragma unroll
  for (int nt = 0; nt < 4; ++nt) {
    const int col = (lane & 15) + nt * 16;
    const float bias = ub1[col];
#pragma unroll
    for (int r = 0; r < 8; ++r)
      ft[(r + hi8) * FF + col] = (_Float16)silu(acc[nt][r] + bias);  // reuse, stride 64
  }

  // u2 = silu(u1 @ uW2 + ub2)
  v8f acc2[4] = {v8f{}, v8f{}, v8f{}, v8f{}};
#pragma unroll
  for (int kb = 0; kb < 2; ++kb) {
    v16h a = load_a_frag(ft, FF, kb, lane);
#pragma unroll
    for (int nt = 0; nt < 4; ++nt)
      acc2[nt] = wmma16(a, load_b_frag_sw(uw2s, 4, kb, nt, lane), acc2[nt]);
  }
#pragma unroll
  for (int nt = 0; nt < 4; ++nt) {
    const int col = (lane & 15) + nt * 16;
    const float bias = ub2[col];
#pragma unroll
    for (int r = 0; r < 8; ++r)
      ft[(r + hi8) * FF + col] = (_Float16)silu(acc2[nt][r] + bias);
  }

  // h += u2 @ uW3 + ub3
  v8f acc3[4] = {v8f{}, v8f{}, v8f{}, v8f{}};
#pragma unroll
  for (int kb = 0; kb < 2; ++kb) {
    v16h a = load_a_frag(ft, FF, kb, lane);
#pragma unroll
    for (int nt = 0; nt < 4; ++nt)
      acc3[nt] = wmma16(a, load_b_frag_sw(uw3s, 4, kb, nt, lane), acc3[nt]);
  }
#pragma unroll
  for (int nt = 0; nt < 4; ++nt) {
    const int col = (lane & 15) + nt * 16;
    const float bias = ub3[col];
#pragma unroll
    for (int r = 0; r < 8; ++r) {
      int i = tileBase + r + hi8;
      if (i < NN) h[(size_t)i * FF + col] += acc3[nt][r] + bias;
    }
  }

  // pos += agg_pm / cnt
  const int base = blockIdx.x * 64;
  for (int idx = tid; idx < 64 * 3; idx += 128) {
    int i = base + idx / 3, c = idx % 3;
    if (i < NN) {
      float rc = __builtin_amdgcn_rcpf(fmaxf(cnt[i], 1.f));
      posw[i * 3 + c] += aggpm[i * 3 + c] * rc;
    }
  }
}

// ---------------- host launch ----------------

extern "C" void kernel_launch(void* const* d_in, const int* in_sizes, int n_in,
                              void* d_out, int out_size, void* d_ws, size_t ws_size,
                              hipStream_t stream) {
  const float* h_in   = (const float*)d_in[0];
  const float* pos    = (const float*)d_in[1];
  const float* cell   = (const float*)d_in[2];
  const int*   esrc   = (const int*)d_in[3];
  const int*   edst   = (const int*)d_in[4];
  const float* emb_W1 = (const float*)d_in[5];
  const float* emb_b1 = (const float*)d_in[6];
  const float* emb_W2 = (const float*)d_in[7];
  const float* emb_b2 = (const float*)d_in[8];
  const float* msg_W1 = (const float*)d_in[9];
  const float* msg_b1 = (const float*)d_in[10];
  const float* bn_g   = (const float*)d_in[11];
  const float* bn_b   = (const float*)d_in[12];
  const float* msg_W2 = (const float*)d_in[13];
  const float* msg_b2 = (const float*)d_in[14];
  const float* upd_W1 = (const float*)d_in[15];
  const float* upd_b1 = (const float*)d_in[16];
  const float* upd_W2 = (const float*)d_in[17];
  const float* upd_b2 = (const float*)d_in[18];
  const float* upd_W3 = (const float*)d_in[19];
  const float* upd_b3 = (const float*)d_in[20];
  const float* pos_W1 = (const float*)d_in[21];
  const float* pos_b1 = (const float*)d_in[22];
  const float* pos_W2 = (const float*)d_in[23];
  const float* pos_b2 = (const float*)d_in[24];
  const float* en_W1  = (const float*)d_in[25];
  const float* en_b1  = (const float*)d_in[26];
  const float* en_W2  = (const float*)d_in[27];
  const float* en_b2  = (const float*)d_in[28];
  const float* en_W3  = (const float*)d_in[29];
  const float* en_b3  = (const float*)d_in[30];
  float* out = (float*)d_out;

  // workspace layout (floats)
  float* ws    = (float*)d_ws;
  float* hbuf  = ws;                               // N*64
  float* posw  = hbuf + (size_t)NN * FF;           // N*3
  float* cnt   = posw + (size_t)NN * 3;            // N
  float* m1    = cnt + NN;                         // E*64
  float* aggm  = m1 + (size_t)NE * FF;             // N*64
  float* aggpm = aggm + (size_t)NN * FF;           // N*3
  float* bnSum = aggpm + (size_t)NN * 3;           // 64
  float* bnSq  = bnSum + FF;                       // 64
  float* bnSc  = bnSq + FF;                        // 64
  float* bnSh  = bnSc + FF;                        // 64
  _Float16* wsw  = (_Float16*)(bnSh + FF);         // NL*WSW_PER_LAYER + WSW_GLOBAL
  _Float16* wswG = wsw + (size_t)NL * WSW_PER_LAYER;

  // init: counts, working pos copy, swizzled weights
  hipMemsetAsync(cnt, 0, (size_t)NN * sizeof(float), stream);
  hipMemcpyAsync(posw, pos, (size_t)NN * 3 * sizeof(float),
                 hipMemcpyDeviceToDevice, stream);
  prep_weights_kernel<<<NL + 1, 256, 0, stream>>>(
      msg_W1, msg_W2, pos_W1, pos_W2, upd_W1, upd_W2, upd_W3,
      emb_W1, emb_W2, en_W1, en_W2, en_W3, wsw);
  count_kernel<<<(NE + 255) / 256, 256, 0, stream>>>(edst, cnt);
  embed_kernel<<<(NN + 63) / 64, 128, 0, stream>>>(h_in, wswG, emb_b1, emb_b2, hbuf);

  const int edgeBlocks = NE / 128;                 // 1250 (32-edge tiles)
  const int nodeBlocks = (NN + 63) / 64;           // 313
  const size_t zeroBytes = ((size_t)NN * FF + (size_t)NN * 3 + 2 * FF) * sizeof(float);

  for (int l = 0; l < NL; ++l) {
    hipMemsetAsync(aggm, 0, zeroBytes, stream);    // agg_m, agg_pm, bnSum, bnSq

    edge_msg1_kernel<<<edgeBlocks, 128, 0, stream>>>(
        hbuf, posw, cell, esrc, edst,
        wsw + (size_t)l * WSW_PER_LAYER, msg_b1 + (size_t)l * FF,
        m1, bnSum, bnSq);

    bn_finalize_kernel<<<1, 64, 0, stream>>>(
        bnSum, bnSq, bn_g + (size_t)l * FF, bn_b + (size_t)l * FF, bnSc, bnSh);

    edge_msg2_kernel<<<edgeBlocks, 128, 0, stream>>>(
        m1, bnSc, bnSh,
        wsw + (size_t)l * WSW_PER_LAYER,
        msg_b2 + (size_t)l * FF, pos_b1 + (size_t)l * FF, pos_b2 + (size_t)l * 3,
        posw, cell, esrc, edst, aggm, aggpm);

    node_update_kernel<<<nodeBlocks, 128, 0, stream>>>(
        hbuf, posw, aggm, aggpm, cnt,
        wsw + (size_t)l * WSW_PER_LAYER,
        upd_b1 + (size_t)l * FF, upd_b2 + (size_t)l * FF, upd_b3 + (size_t)l * FF);
  }

  energy_kernel<<<(NN + 63) / 64, 128, 0, stream>>>(
      hbuf, wswG, en_b1, en_b2, en_b3, out);
}
